// NormalMoeExpertsCPUInfer_17867063951969
// MI455X (gfx1250) — matrix operand order repaired and checked by reading, not compiled, not run on hardware
//
#include <hip/hip_runtime.h>
#include <hip/hip_bf16.h>

#define T_TOK 2048
#define KSEL  6
#define DIM   2048
#define INTER 1408
#define NEXP  16

typedef __attribute__((ext_vector_type(16))) _Float16 v16h;
typedef __attribute__((ext_vector_type(8)))  _Float16 v8h;
typedef __attribute__((ext_vector_type(4)))  _Float16 v4h;
typedef __attribute__((ext_vector_type(8)))  float    v8f;

// LDS tile row stride in halves: 32 payload + 8 pad -> 80B row stride, conflict-free b128 reads
#define LDS_STRIDE 40

__device__ __forceinline__ v16h frag_pack(const _Float16* lo_p, const _Float16* hi_p) {
    v8h lo = *(const v8h*)lo_p;
    v8h hi = *(const v8h*)hi_p;
    v16h r;
#pragma unroll
    for (int j = 0; j < 8; ++j) { r[j] = lo[j]; r[8 + j] = hi[j]; }
    return r;
}

// A fragment (16x32 f16, M x K) loaded directly from row-major f16 global memory.
// ISA layout: lanes 0-15 M=lane, K in {0..7,16..23}; lanes 16-31 M=lane-16, K in {8..15,24..31}
__device__ __forceinline__ v16h load_a_global(const _Float16* __restrict__ base, int ld,
                                              int row, int kb, int lane) {
    const _Float16* q = base + (size_t)(row + (lane & 15)) * ld + kb + ((lane >> 4) & 1) * 8;
    return frag_pack(q, q + 16);
}

// B fragment (32x16 f16, K x N) from an LDS tile holding W rows (N-major, K contiguous).
// Layout: lanes 0-15 hold K in {0..7,16..23}, lanes 16-31 hold K in {8..15,24..31}, N=lane&15
__device__ __forceinline__ v16h load_b_lds(const _Float16* tile, int n, int lane) {
    const _Float16* q = tile + (size_t)(n + (lane & 15)) * LDS_STRIDE + ((lane >> 4) & 1) * 8;
    return frag_pack(q, q + 16);
}

__device__ __forceinline__ v8f wmma_acc(v16h a, v16h b, v8f c) {
    return __builtin_amdgcn_wmma_f32_16x16x32_f16(false, a, false, b, (short)0, c, false, false);
}

// fast silu: g * rcp(1 + exp(-g))  -> v_exp_f32 + v_rcp_f32, no IEEE div expansion
__device__ __forceinline__ float fast_silu(float g) {
    return g * __builtin_amdgcn_rcpf(1.0f + __expf(-g));
}

// Stage a 64x32 f32 weight tile -> f16 LDS tile (256 threads, 8 elements each)
__device__ __forceinline__ void stage_w_tile(_Float16* tile, const float* __restrict__ src,
                                             int ld, int row0, int kb, int tid) {
    const int r  = tid >> 2;          // 0..63
    const int kq = (tid & 3) << 3;    // 0,8,16,24
    const float* s = src + (size_t)(row0 + r) * ld + kb + kq;
    float4 f0 = *(const float4*)(s);
    float4 f1 = *(const float4*)(s + 4);
    v8h h;
    h[0] = (_Float16)f0.x; h[1] = (_Float16)f0.y; h[2] = (_Float16)f0.z; h[3] = (_Float16)f0.w;
    h[4] = (_Float16)f1.x; h[5] = (_Float16)f1.y; h[6] = (_Float16)f1.z; h[7] = (_Float16)f1.w;
    *(v8h*)&tile[(size_t)r * LDS_STRIDE + kq] = h;
}

// ---------------- prep kernels ----------------

__global__ void zero_f32(float* __restrict__ y, int n) {
    int i = blockIdx.x * 256 + threadIdx.x;
    if (i < n) y[i] = 0.0f;
}

__global__ void gate_combine(const float* __restrict__ w, const int* __restrict__ idx,
                             float* __restrict__ gate) {
    int t = blockIdx.x * 256 + threadIdx.x;
    if (t >= T_TOK) return;
    float acc[NEXP];
#pragma unroll
    for (int e = 0; e < NEXP; ++e) acc[e] = 0.0f;
#pragma unroll
    for (int k = 0; k < KSEL; ++k) {
        int e = idx[t * KSEL + k] & (NEXP - 1);
        acc[e] += w[t * KSEL + k];
    }
#pragma unroll
    for (int e = 0; e < NEXP; ++e) gate[(size_t)t * NEXP + e] = acc[e];
}

__global__ void cvt_x_f16(const float* __restrict__ x, _Float16* __restrict__ xh) {
    size_t i = ((size_t)blockIdx.x * 256 + threadIdx.x) * 4;
    float4 f = *(const float4*)(x + i);
    v4h h;
    h[0] = (_Float16)f.x; h[1] = (_Float16)f.y; h[2] = (_Float16)f.z; h[3] = (_Float16)f.w;
    *(v4h*)(xh + i) = h;
}

// ---------------- expert GEMM 1: act = silu(x@Wg^T) * (x@Wu^T) * gate ----------------
// block tile 128 (tokens) x 64 (inter), 8 waves, each wave owns 32x32.
// K staged 64 per barrier pair (two 32-K sub-steps) -> 16 WMMA per wave per barrier.

__global__ __launch_bounds__(256) void moe_gate_up(
    const float* __restrict__ Wg, const float* __restrict__ Wu,
    const _Float16* __restrict__ xh, const float* __restrict__ gate,
    _Float16* __restrict__ act, int e)
{
    __shared__ __align__(16) _Float16 lw[2][2][64 * LDS_STRIDE];  // [gate|up][ksub]
    __shared__ float Gs[128];

    const int tid  = threadIdx.x;
    const int lane = tid & 31;
    const int wave = tid >> 5;
    const int wm = wave >> 1;   // 0..3  (32 rows each)
    const int wn = wave & 1;    // 0..1  (32 cols each)
    const int n0 = blockIdx.x * 64;
    const int t0 = blockIdx.y * 128;

    if (tid < 128) Gs[tid] = gate[(size_t)(t0 + tid) * NEXP + e];

    const float* wgp = Wg + (size_t)e * INTER * DIM;
    const float* wup = Wu + (size_t)e * INTER * DIM;

    v8f cg[2][2], cu[2][2];
    const v8f vz = {};
#pragma unroll
    for (int i = 0; i < 2; ++i)
#pragma unroll
        for (int j = 0; j < 2; ++j) { cg[i][j] = vz; cu[i][j] = vz; }

    const int tRow = t0 + wm * 32;

    for (int kb = 0; kb < DIM; kb += 64) {
        __syncthreads();
        stage_w_tile(&lw[0][0][0], wgp, DIM, n0, kb,      tid);
        stage_w_tile(&lw[0][1][0], wgp, DIM, n0, kb + 32, tid);
        stage_w_tile(&lw[1][0][0], wup, DIM, n0, kb,      tid);
        stage_w_tile(&lw[1][1][0], wup, DIM, n0, kb + 32, tid);
        __syncthreads();

#pragma unroll
        for (int ks = 0; ks < 2; ++ks) {
            const int kk = kb + ks * 32;
            v16h a0 = load_a_global(xh, DIM, tRow,      kk, lane);
            v16h a1 = load_a_global(xh, DIM, tRow + 16, kk, lane);
            v16h bg0 = load_b_lds(&lw[0][ks][0], wn * 32,      lane);
            v16h bg1 = load_b_lds(&lw[0][ks][0], wn * 32 + 16, lane);
            v16h bu0 = load_b_lds(&lw[1][ks][0], wn * 32,      lane);
            v16h bu1 = load_b_lds(&lw[1][ks][0], wn * 32 + 16, lane);

            cg[0][0] = wmma_acc(a0, bg0, cg[0][0]);
            cg[0][1] = wmma_acc(a0, bg1, cg[0][1]);
            cg[1][0] = wmma_acc(a1, bg0, cg[1][0]);
            cg[1][1] = wmma_acc(a1, bg1, cg[1][1]);
            cu[0][0] = wmma_acc(a0, bu0, cu[0][0]);
            cu[0][1] = wmma_acc(a0, bu1, cu[0][1]);
            cu[1][0] = wmma_acc(a1, bu0, cu[1][0]);
            cu[1][1] = wmma_acc(a1, bu1, cu[1][1]);
        }
    }

    // epilogue: act[t, n] = silu(g) * u * gate_w
    const int lhi = (lane >> 4) & 1;
    const int lc  = lane & 15;
#pragma unroll
    for (int ms = 0; ms < 2; ++ms) {
#pragma unroll
        for (int r = 0; r < 8; ++r) {
            const int rl = wm * 32 + ms * 16 + lhi * 8 + r;
            const float gw = Gs[rl];
            const size_t t = (size_t)(t0 + rl);
#pragma unroll
            for (int ns = 0; ns < 2; ++ns) {
                float g = cg[ms][ns][r];
                float u = cu[ms][ns][r];
                float v = fast_silu(g) * u * gw;
                act[t * INTER + n0 + wn * 32 + ns * 16 + lc] = (_Float16)v;
            }
        }
    }
}

// ---------------- expert GEMM 2: y += act @ Wd^T ----------------
// block tile 128 (tokens) x 64 (dim), K over INTER staged 64 per barrier pair

__global__ __launch_bounds__(256) void moe_down(
    const float* __restrict__ Wd, const _Float16* __restrict__ acth,
    float* __restrict__ y, int e)
{
    __shared__ __align__(16) _Float16 lw[2][64 * LDS_STRIDE];   // [ksub]

    const int tid  = threadIdx.x;
    const int lane = tid & 31;
    const int wave = tid >> 5;
    const int wm = wave >> 1;
    const int wn = wave & 1;
    const int d0 = blockIdx.x * 64;
    const int t0 = blockIdx.y * 128;

    const float* wdp = Wd + (size_t)e * DIM * INTER;

    v8f c[2][2];
    const v8f vz = {};
#pragma unroll
    for (int i = 0; i < 2; ++i)
#pragma unroll
        for (int j = 0; j < 2; ++j) c[i][j] = vz;

    const int tRow = t0 + wm * 32;

    for (int kb = 0; kb < INTER; kb += 64) {
        __syncthreads();
        stage_w_tile(&lw[0][0], wdp, INTER, d0, kb,      tid);
        stage_w_tile(&lw[1][0], wdp, INTER, d0, kb + 32, tid);
        __syncthreads();

#pragma unroll
        for (int ks = 0; ks < 2; ++ks) {
            const int kk = kb + ks * 32;
            v16h a0 = load_a_global(acth, INTER, tRow,      kk, lane);
            v16h a1 = load_a_global(acth, INTER, tRow + 16, kk, lane);
            v16h b0 = load_b_lds(&lw[ks][0], wn * 32,      lane);
            v16h b1 = load_b_lds(&lw[ks][0], wn * 32 + 16, lane);

            c[0][0] = wmma_acc(a0, b0, c[0][0]);
            c[0][1] = wmma_acc(a0, b1, c[0][1]);
            c[1][0] = wmma_acc(a1, b0, c[1][0]);
            c[1][1] = wmma_acc(a1, b1, c[1][1]);
        }
    }

    const int lhi = (lane >> 4) & 1;
    const int lc  = lane & 15;
#pragma unroll
    for (int ms = 0; ms < 2; ++ms) {
#pragma unroll
        for (int r = 0; r < 8; ++r) {
            const size_t t = (size_t)(t0 + wm * 32 + ms * 16 + lhi * 8 + r);
#pragma unroll
            for (int ns = 0; ns < 2; ++ns) {
                const int d = d0 + wn * 32 + ns * 16 + lc;
                y[t * DIM + d] += c[ms][ns][r];   // experts run sequentially -> no race
            }
        }
    }
}

extern "C" void kernel_launch(void* const* d_in, const int* in_sizes, int n_in,
                              void* d_out, int out_size, void* d_ws, size_t ws_size,
                              hipStream_t stream) {
    const float* x   = (const float*)d_in[0];
    const float* wts = (const float*)d_in[1];
    const int*   idx = (const int*)d_in[2];
    const float* Wg  = (const float*)d_in[3];
    const float* Wu  = (const float*)d_in[4];
    const float* Wd  = (const float*)d_in[5];
    float* y = (float*)d_out;

    char* ws = (char*)d_ws;
    float*    gate = (float*)ws;                                   // T*E f32   = 128 KB
    _Float16* xh   = (_Float16*)(ws + (size_t)T_TOK * NEXP * 4);   // T*DIM f16 = 8 MB
    _Float16* acth = (_Float16*)(ws + (size_t)T_TOK * NEXP * 4
                                    + (size_t)T_TOK * DIM * 2);    // T*INTER f16 = 5.5 MB

    zero_f32<<<(T_TOK * DIM + 255) / 256, 256, 0, stream>>>(y, T_TOK * DIM);
    gate_combine<<<(T_TOK + 255) / 256, 256, 0, stream>>>(wts, idx, gate);
    cvt_x_f16<<<((size_t)T_TOK * DIM / 4 + 255) / 256, 256, 0, stream>>>(x, xh);

    dim3 g1(INTER / 64, T_TOK / 128);   // 22 x 16
    dim3 g2(DIM / 64,   T_TOK / 128);   // 32 x 16
    for (int e = 0; e < NEXP; ++e) {
        moe_gate_up<<<g1, 256, 0, stream>>>(Wg, Wu, xh, gate, acth, e);
        moe_down<<<g2, 256, 0, stream>>>(Wd, acth, y, e);
    }
}
